// Graph2Graph_18047452578376
// MI455X (gfx1250) — matrix operand order repaired and checked by reading, not compiled, not run on hardware
//
#include <hip/hip_runtime.h>
#include <hip/hip_bf16.h>

// ---------------------------------------------------------------------------
// MI455X (gfx1250) implementation of the Graph2Graph block.
// Hot path: per-edge / per-node FCNNs via v_wmma_f32_16x16x32_f16 (f16 inputs,
// f32 accumulate), weights resident in LDS, f32 everywhere else.
// ---------------------------------------------------------------------------

typedef __attribute__((ext_vector_type(16))) _Float16 v16h;
typedef __attribute__((ext_vector_type(8)))  float    v8f;

union V16H { v16h v; uint4 q[2]; _Float16 e[16]; };
union V8F  { v8f  v; float e[8]; };

#define WAIT_DS() asm volatile("s_wait_dscnt 0" ::: "memory")

// A operand (16x32 f16, rows striped in-lane):  lane<16 -> row=lane, K=k0+{0..7,16..23}
//                                               lane>=16 -> row=lane-16, K=k0+{8..15,24..31}
__device__ __forceinline__ v16h load_a(const _Float16* p, int lane, int rowStride, int k0) {
  int r  = lane & 15;
  int kb = k0 + ((lane >> 4) << 3);
  const _Float16* row = p + r * rowStride;
  V16H a;
  a.q[0] = *(const uint4*)(row + kb);
  a.q[1] = *(const uint4*)(row + kb + 16);
  return a.v;
}

// B operand (32x16 f16): lane<16 -> col=lane, K=k0+{0..7,16..23} ; lane>=16 -> col=lane-16, K=k0+{8..15,24..31}
// Weights stored transposed [N][K] so this is two contiguous b128 reads.
__device__ __forceinline__ v16h load_b(const _Float16* wT, int lane, int rowStride, int n0, int k0) {
  int n  = n0 + (lane & 15);
  int kb = k0 + ((lane >> 4) << 3);
  const _Float16* row = wT + n * rowStride;
  V16H b;
  b.q[0] = *(const uint4*)(row + kb);
  b.q[1] = *(const uint4*)(row + kb + 16);
  return b.v;
}

__device__ __forceinline__ v8f wmma_f16(v16h a, v16h b, v8f c) {
  return __builtin_amdgcn_wmma_f32_16x16x32_f16(false, a, false, b, (short)0, c, false, false);
}

// ---------------------------------------------------------------------------
// K1: convert + transpose one weight set to f16:  w1(3,64,256)->w1T(3,256,64),
//     w2(3,256,64)->w2T(3,64,256)
// ---------------------------------------------------------------------------
__global__ void cvt_w_kernel(const float* __restrict__ w1, const float* __restrict__ w2,
                             _Float16* __restrict__ w1T, _Float16* __restrict__ w2T) {
  int i = blockIdx.x * blockDim.x + threadIdx.x;          // over 3*64*256 == 3*256*64
  if (i < 3 * 64 * 256) {
    int f = i / 16384, rem = i % 16384;
    int k = rem / 256, n = rem % 256;
    w1T[f * 16384 + n * 64 + k] = (_Float16)w1[i];
    int j = rem / 64, n2 = rem % 64;
    w2T[f * 16384 + n2 * 256 + j] = (_Float16)w2[i];
  }
}

// ---------------------------------------------------------------------------
// K2: three g0-FCNNs (slice f=2 of e/n/g weight sets) + fold output biases.
//   b2e_sum = e_b2[0]+e_b2[1]+FCNN_e2(g0)
//   hxbias  = n_b2[0]+FCNN_n2(g0)
//   ggb     = g_b2[0]+g_b2[1]+FCNN_g2(g0)
// ---------------------------------------------------------------------------
__global__ __launch_bounds__(256) void gvec_kernel(
    const float* g0,
    const float* ew1, const float* eb1, const float* ew2, const float* eb2,
    const float* nw1, const float* nb1, const float* nw2, const float* nb2,
    const float* gw1, const float* gb1, const float* gw2, const float* gb2,
    float* b2e_sum, float* hxbias, float* ggb) {
  __shared__ float g0s[64], h[256], outv[64];
  int t = threadIdx.x;
  if (t < 64) g0s[t] = g0[t];
  __syncthreads();
  const float* W1[3] = {ew1 + 32768, nw1 + 32768, gw1 + 32768};
  const float* B1[3] = {eb1 + 512,   nb1 + 512,   gb1 + 512};
  const float* W2[3] = {ew2 + 32768, nw2 + 32768, gw2 + 32768};
  const float* B2[3] = {eb2 + 128,   nb2 + 128,   gb2 + 128};
  for (int s = 0; s < 3; ++s) {
    float a = B1[s][t];
    for (int k = 0; k < 64; ++k) a += g0s[k] * W1[s][k * 256 + t];
    h[t] = a > 0.f ? a : 0.f;
    __syncthreads();
    if (t < 64) {
      float o = B2[s][t];
      for (int j = 0; j < 256; ++j) o += h[j] * W2[s][j * 64 + t];
      outv[t] = o;
    }
    __syncthreads();
    if (t < 64) {
      if (s == 0)      b2e_sum[t] = eb2[t] + eb2[64 + t] + outv[t];
      else if (s == 1) hxbias[t]  = nb2[t] + outv[t];
      else             ggb[t]     = outv[t] + gb2[t] + gb2[64 + t];
    }
    __syncthreads();
  }
}

// ---------------------------------------------------------------------------
// K3: init scatter buffers
// ---------------------------------------------------------------------------
__global__ void init_ws_kernel(float* denom, float* agg, float* xsum, float* esum, long n) {
  long i = (long)blockIdx.x * blockDim.x + threadIdx.x;
  long stride = (long)gridDim.x * blockDim.x;
  for (; i < n; i += stride) { denom[i] = 1e-7f; agg[i] = 0.f; }
  if (blockIdx.x == 0 && threadIdx.x < 64) { xsum[threadIdx.x] = 0.f; esum[threadIdx.x] = 0.f; }
}

// ---------------------------------------------------------------------------
// K4: node GEMMs: hx = FCNN0(x0)+gn-bias ; fx = FCNN1(x0)+b2[1].
// LDS: 128KB weights + 8 waves * 3KB staging. One 16-row tile per wave.
// ---------------------------------------------------------------------------
__global__ __launch_bounds__(256) void node_gemm_kernel(
    const float* __restrict__ x0, const _Float16* __restrict__ w1T,
    const _Float16* __restrict__ w2T, const float* __restrict__ b1,
    const float* __restrict__ bias0, const float* __restrict__ bias1,
    float* __restrict__ hx, float* __restrict__ fx, int N) {
  extern __shared__ char smem[];
  _Float16* sW1 = (_Float16*)smem;                 // 2*256*64
  _Float16* sW2 = sW1 + 2 * 256 * 64;              // 2*64*256
  int wave = threadIdx.x >> 5, lane = threadIdx.x & 31;
  char* pw = smem + 131072 + wave * 3072;
  _Float16* A   = (_Float16*)pw;                   // 16x64
  _Float16* Hst = A + 16 * 64;                     // 16x32

  { // weights -> LDS
    const uint4* g1 = (const uint4*)w1T; uint4* s1 = (uint4*)sW1;
    for (int i = threadIdx.x; i < 4096; i += 256) s1[i] = g1[i];
    const uint4* g2 = (const uint4*)w2T; uint4* s2 = (uint4*)sW2;
    for (int i = threadIdx.x; i < 4096; i += 256) s2[i] = g2[i];
  }
  __syncthreads();

  int nTiles = N >> 4;
  int nwaves = gridDim.x * 8;
  int nl = lane & 15, m0 = (lane >> 4) << 3;
  for (int tile = blockIdx.x * 8 + wave; tile < nTiles; tile += nwaves) {
    int base = tile << 4;
    { // stage A (f16) : 2 lanes per row
      int r = lane >> 1, c0 = (lane & 1) << 5;
      const float4* xr = (const float4*)(x0 + (long)(base + r) * 64 + c0);
      _Float16* ar = A + r * 64 + c0;
      #pragma unroll
      for (int j = 0; j < 8; ++j) {
        float4 q = xr[j];
        ar[j * 4 + 0] = (_Float16)q.x; ar[j * 4 + 1] = (_Float16)q.y;
        ar[j * 4 + 2] = (_Float16)q.z; ar[j * 4 + 3] = (_Float16)q.w;
      }
    }
    WAIT_DS();
    v8f acc[2][4];
    #pragma unroll
    for (int nt = 0; nt < 4; ++nt) {
      float b0v = bias0[nt * 16 + nl], b1v = bias1[nt * 16 + nl];
      acc[0][nt] = (v8f){b0v, b0v, b0v, b0v, b0v, b0v, b0v, b0v};
      acc[1][nt] = (v8f){b1v, b1v, b1v, b1v, b1v, b1v, b1v, b1v};
    }
    for (int f = 0; f < 2; ++f) {
      const _Float16* W1 = sW1 + f * 16384;
      const _Float16* W2 = sW2 + f * 16384;
      const float* b1f = b1 + f * 256;
      for (int hc = 0; hc < 8; ++hc) {
        #pragma unroll
        for (int sub = 0; sub < 2; ++sub) {
          int n0 = hc * 32 + sub * 16;
          float hb = b1f[n0 + nl];
          v8f h = (v8f){hb, hb, hb, hb, hb, hb, hb, hb};
          #pragma unroll
          for (int k0 = 0; k0 < 64; k0 += 32)
            h = wmma_f16(load_a(A, lane, 64, k0), load_b(W1, lane, 64, n0, k0), h);
          V8F hv; hv.v = h;
          #pragma unroll
          for (int i = 0; i < 8; ++i) {
            float tt = hv.e[i]; tt = tt > 0.f ? tt : 0.f;
            Hst[(m0 + i) * 32 + sub * 16 + nl] = (_Float16)tt;
          }
        }
        WAIT_DS();
        v16h a2 = load_a(Hst, lane, 32, 0);
        #pragma unroll
        for (int nt = 0; nt < 4; ++nt)
          acc[f][nt] = wmma_f16(a2, load_b(W2, lane, 256, nt * 16, hc * 32), acc[f][nt]);
        WAIT_DS();
      }
    }
    // scatter C-layout to global
    #pragma unroll
    for (int nt = 0; nt < 4; ++nt) {
      V8F a0; a0.v = acc[0][nt]; V8F a1; a1.v = acc[1][nt];
      #pragma unroll
      for (int i = 0; i < 8; ++i) {
        long idx = (long)(base + m0 + i) * 64 + nt * 16 + nl;
        hx[idx] = a0.e[i];
        fx[idx] = a1.e[i];
      }
    }
  }
}

// ---------------------------------------------------------------------------
// K5: edge block. Per 16-edge tile: he = FCNN0(x[u]+x[v]) + FCNN1(e0) + ge
// (+all b2 folded); InstanceNorm rows; e1 = e0 + relu(norm); sigmoid gates and
// 4-way scatter atomics; e1 column sums in registers.
// ---------------------------------------------------------------------------
__global__ __launch_bounds__(256) void edge_kernel(
    const float* __restrict__ x0, const int* __restrict__ ei,
    const float* __restrict__ e0, const _Float16* __restrict__ w1T,
    const _Float16* __restrict__ w2T, const float* __restrict__ b1,
    const float* __restrict__ b2sum, const float* __restrict__ fx,
    float* __restrict__ e1_out, float* __restrict__ denom,
    float* __restrict__ agg, float* __restrict__ esum, int E) {
  extern __shared__ char smem[];
  _Float16* sW1 = (_Float16*)smem;
  _Float16* sW2 = sW1 + 2 * 256 * 64;
  int wave = threadIdx.x >> 5, lane = threadIdx.x & 31;
  char* pw = smem + 131072 + wave * 9216;
  _Float16* Ax  = (_Float16*)pw;           // 16x64 f16
  _Float16* Ae  = Ax + 16 * 64;            // 16x64 f16
  _Float16* Hst = Ae + 16 * 64;            // 16x32 f16
  float*    Ost = (float*)(pw + 5120);     // 16x64 f32

  { // weights -> LDS (128 KB)
    const uint4* g1 = (const uint4*)w1T; uint4* s1 = (uint4*)sW1;
    for (int i = threadIdx.x; i < 4096; i += 256) s1[i] = g1[i];
    const uint4* g2 = (const uint4*)w2T; uint4* s2 = (uint4*)sW2;
    for (int i = threadIdx.x; i < 4096; i += 256) s2[i] = g2[i];
  }
  __syncthreads();

  int nTiles = E >> 4;
  int nwaves = gridDim.x * 8;
  int nl = lane & 15, m0 = (lane >> 4) << 3;
  float es0 = 0.f, es1 = 0.f;

  for (int tile = blockIdx.x * 8 + wave; tile < nTiles; tile += nwaves) {
    int ebase = tile << 4;
    { // stage gathered A tiles
      int r = lane >> 1, c0 = (lane & 1) << 5;
      int e = ebase + r;
      int u = ei[e], v = ei[E + e];
      const float4* xu = (const float4*)(x0 + (long)u * 64 + c0);
      const float4* xv = (const float4*)(x0 + (long)v * 64 + c0);
      const float4* er = (const float4*)(e0 + (long)e * 64 + c0);
      _Float16* axr = Ax + r * 64 + c0;
      _Float16* aer = Ae + r * 64 + c0;
      #pragma unroll
      for (int j = 0; j < 8; ++j) {
        float4 a = xu[j], b = xv[j], c = er[j];
        axr[j * 4 + 0] = (_Float16)(a.x + b.x); axr[j * 4 + 1] = (_Float16)(a.y + b.y);
        axr[j * 4 + 2] = (_Float16)(a.z + b.z); axr[j * 4 + 3] = (_Float16)(a.w + b.w);
        aer[j * 4 + 0] = (_Float16)c.x; aer[j * 4 + 1] = (_Float16)c.y;
        aer[j * 4 + 2] = (_Float16)c.z; aer[j * 4 + 3] = (_Float16)c.w;
      }
      // prefetch next tile's edge features (-> global_prefetch_b8)
      long pnext = (long)(ebase + (nwaves << 4)) * 64;
      if (pnext < (long)E * 64) __builtin_prefetch(e0 + pnext + lane * 32, 0, 1);
    }
    WAIT_DS();

    v8f acc[4];
    #pragma unroll
    for (int nt = 0; nt < 4; ++nt) {
      float bv = b2sum[nt * 16 + nl];
      acc[nt] = (v8f){bv, bv, bv, bv, bv, bv, bv, bv};
    }
    for (int f = 0; f < 2; ++f) {
      const _Float16* Aop = f ? Ae : Ax;
      const _Float16* W1 = sW1 + f * 16384;
      const _Float16* W2 = sW2 + f * 16384;
      const float* b1f = b1 + f * 256;
      for (int hc = 0; hc < 8; ++hc) {
        #pragma unroll
        for (int sub = 0; sub < 2; ++sub) {
          int n0 = hc * 32 + sub * 16;
          float hb = b1f[n0 + nl];
          v8f h = (v8f){hb, hb, hb, hb, hb, hb, hb, hb};
          #pragma unroll
          for (int k0 = 0; k0 < 64; k0 += 32)
            h = wmma_f16(load_a(Aop, lane, 64, k0), load_b(W1, lane, 64, n0, k0), h);
          V8F hv; hv.v = h;
          #pragma unroll
          for (int i = 0; i < 8; ++i) {
            float tt = hv.e[i]; tt = tt > 0.f ? tt : 0.f;
            Hst[(m0 + i) * 32 + sub * 16 + nl] = (_Float16)tt;
          }
        }
        WAIT_DS();
        v16h a2 = load_a(Hst, lane, 32, 0);
        #pragma unroll
        for (int nt = 0; nt < 4; ++nt)
          acc[nt] = wmma_f16(a2, load_b(W2, lane, 256, nt * 16, hc * 32), acc[nt]);
        WAIT_DS();
      }
    }

    // he tile -> LDS for row-wise InstanceNorm
    #pragma unroll
    for (int nt = 0; nt < 4; ++nt) {
      V8F av; av.v = acc[nt];
      #pragma unroll
      for (int i = 0; i < 8; ++i) Ost[(m0 + i) * 64 + nt * 16 + nl] = av.e[i];
    }
    WAIT_DS();

    if (lane < 16) {
      int r = lane, e = ebase + r;
      const float* he = Ost + r * 64;
      float s = 0.f, ss = 0.f;
      for (int c = 0; c < 64; ++c) { float x = he[c]; s += x; ss += x * x; }
      float m = s * 0.015625f;
      float var = ss * 0.015625f - m * m;
      float rs = rsqrtf(var + 1e-5f);
      int u = ei[e], v = ei[E + e];
      const float* e0r = e0 + (long)e * 64;
      float* e1r = e1_out + (long)e * 64;
      const float* fxu = fx + (long)u * 64;
      const float* fxv = fx + (long)v * 64;
      float* du = denom + (long)u * 64; float* dv = denom + (long)v * 64;
      float* au = agg + (long)u * 64;   float* av_ = agg + (long)v * 64;
      for (int c = 0; c < 64; ++c) {
        float nh = (he[c] - m) * rs;
        float e1 = e0r[c] + (nh > 0.f ? nh : 0.f);
        e1r[c] = e1;
        float se = 1.f / (1.f + __expf(-e1));
        atomicAdd(du + c, se);
        atomicAdd(dv + c, se);
        atomicAdd(au + c, se * fxv[c]);
        atomicAdd(av_ + c, se * fxu[c]);
        Ost[r * 64 + c] = e1;       // reuse tile for column sums
      }
    }
    WAIT_DS();
    { // per-tile column partials -> wave-local registers
      int c = lane << 1;
      float p0 = 0.f, p1 = 0.f;
      #pragma unroll
      for (int r = 0; r < 16; ++r) { p0 += Ost[r * 64 + c]; p1 += Ost[r * 64 + c + 1]; }
      es0 += p0; es1 += p1;
    }
  }
  atomicAdd(esum + (lane << 1), es0);
  atomicAdd(esum + (lane << 1) + 1, es1);
}

// ---------------------------------------------------------------------------
// K6: node finalize: hn = hx + agg/denom (bias incl. gn), norm, residual, xsum.
// ---------------------------------------------------------------------------
__global__ __launch_bounds__(256) void node_fin_kernel(
    const float* __restrict__ x0, const float* __restrict__ hx,
    const float* __restrict__ agg, const float* __restrict__ denom,
    float* __restrict__ x1_out, float* __restrict__ xsum, int N) {
  __shared__ float Ost[8][16 * 64];
  int wave = threadIdx.x >> 5, lane = threadIdx.x & 31;
  float s0 = 0.f, s1 = 0.f;
  int nT = N >> 4;
  for (int tile = blockIdx.x * 8 + wave; tile < nT; tile += gridDim.x * 8) {
    int base = tile << 4;
    if (lane < 16) {
      long i = base + lane;
      const float* hr = hx + i * 64;
      const float* ar = agg + i * 64;
      const float* dr = denom + i * 64;
      float s = 0.f, ss = 0.f;
      for (int c = 0; c < 64; ++c) {
        float h = hr[c] + ar[c] / dr[c];
        s += h; ss += h * h;
      }
      float m = s * 0.015625f;
      float var = ss * 0.015625f - m * m;
      float rs = rsqrtf(var + 1e-5f);
      const float* x0r = x0 + i * 64;
      float* xo = x1_out + i * 64;
      for (int c = 0; c < 64; ++c) {
        float h = hr[c] + ar[c] / dr[c];
        float nh = (h - m) * rs;
        float x1 = x0r[c] + (nh > 0.f ? nh : 0.f);
        xo[c] = x1;
        Ost[wave][lane * 64 + c] = x1;
      }
    }
    WAIT_DS();
    int c = lane << 1;
    float p0 = 0.f, p1 = 0.f;
    #pragma unroll
    for (int r = 0; r < 16; ++r) { p0 += Ost[wave][r * 64 + c]; p1 += Ost[wave][r * 64 + c + 1]; }
    s0 += p0; s1 += p1;
  }
  atomicAdd(xsum + (lane << 1), s0);
  atomicAdd(xsum + (lane << 1) + 1, s1);
}

// ---------------------------------------------------------------------------
// K7: global finalize.
// ---------------------------------------------------------------------------
__global__ __launch_bounds__(256) void gfin_kernel(
    const float* g0, const float* xsum, const float* esum,
    const float* gw1, const float* gb1, const float* gw2, const float* ggb,
    float* gout, float invN, float invE) {
  __shared__ float xm[64], em[64], h[256], outv[64], stats[2];
  int t = threadIdx.x;
  if (t < 64) { xm[t] = xsum[t] * invN; em[t] = esum[t] * invE; outv[t] = ggb[t]; }
  __syncthreads();
  for (int s = 0; s < 2; ++s) {
    const float* in = s ? em : xm;
    const float* W1 = gw1 + s * 16384;
    const float* B1 = gb1 + s * 256;
    const float* W2 = gw2 + s * 16384;
    float a = B1[t];
    for (int k = 0; k < 64; ++k) a += in[k] * W1[k * 256 + t];
    h[t] = a > 0.f ? a : 0.f;
    __syncthreads();
    if (t < 64) {
      float o = 0.f;
      for (int j = 0; j < 256; ++j) o += h[j] * W2[j * 64 + t];
      outv[t] += o;
    }
    __syncthreads();
  }
  if (t == 0) {
    float s = 0.f, ss = 0.f;
    for (int c = 0; c < 64; ++c) { s += outv[c]; ss += outv[c] * outv[c]; }
    float m = s * 0.015625f;
    float var = ss * 0.015625f - m * m;
    stats[0] = m; stats[1] = rsqrtf(var + 1e-5f);
  }
  __syncthreads();
  if (t < 64) {
    float nh = (outv[t] - stats[0]) * stats[1];
    gout[t] = g0[t] + (nh > 0.f ? nh : 0.f);
  }
}

// ---------------------------------------------------------------------------
extern "C" void kernel_launch(void* const* d_in, const int* in_sizes, int n_in,
                              void* d_out, int out_size, void* d_ws, size_t ws_size,
                              hipStream_t stream) {
  const float* x0 = (const float*)d_in[0];
  const int*   ei = (const int*)d_in[1];
  const float* e0 = (const float*)d_in[2];
  const float* g0 = (const float*)d_in[3];
  const float *ew1 = (const float*)d_in[4],  *eb1 = (const float*)d_in[5];
  const float *ew2 = (const float*)d_in[6],  *eb2 = (const float*)d_in[7];
  const float *nw1 = (const float*)d_in[8],  *nb1 = (const float*)d_in[9];
  const float *nw2 = (const float*)d_in[10], *nb2 = (const float*)d_in[11];
  const float *gw1 = (const float*)d_in[12], *gb1 = (const float*)d_in[13];
  const float *gw2 = (const float*)d_in[14], *gb2 = (const float*)d_in[15];

  int N = in_sizes[0] / 64;
  int E = in_sizes[2] / 64;

  float* x1 = (float*)d_out;
  float* e1 = x1 + (long)N * 64;
  float* g1 = e1 + (long)E * 64;

  // workspace layout
  char* ws = (char*)d_ws;
  _Float16* eW1T = (_Float16*)ws;
  _Float16* eW2T = eW1T + 49152;
  _Float16* nW1T = eW2T + 49152;
  _Float16* nW2T = nW1T + 49152;
  float* b2e  = (float*)(ws + 393216);
  float* hxb  = b2e + 64;
  float* ggb  = hxb + 64;
  float* xsum = ggb + 64;
  float* esum = xsum + 64;
  float* hx    = esum + 64;
  float* fx    = hx + (long)N * 64;
  float* denom = fx + (long)N * 64;
  float* agg   = denom + (long)N * 64;

  const int edgeLds = 131072 + 8 * 9216;   // 204800 B
  const int nodeLds = 131072 + 8 * 3072;   // 155648 B
  hipFuncSetAttribute(reinterpret_cast<const void*>(edge_kernel),
                      hipFuncAttributeMaxDynamicSharedMemorySize, edgeLds);
  hipFuncSetAttribute(reinterpret_cast<const void*>(node_gemm_kernel),
                      hipFuncAttributeMaxDynamicSharedMemorySize, nodeLds);

  // K1: weight convert/transpose (e and n sets)
  cvt_w_kernel<<<192, 256, 0, stream>>>(ew1, ew2, eW1T, eW2T);
  cvt_w_kernel<<<192, 256, 0, stream>>>(nw1, nw2, nW1T, nW2T);
  // K2: g0 FCNNs + folded biases
  gvec_kernel<<<1, 256, 0, stream>>>(g0, ew1, eb1, ew2, eb2, nw1, nb1, nw2, nb2,
                                     gw1, gb1, gw2, gb2, b2e, hxb, ggb);
  // K3: init scatter buffers
  init_ws_kernel<<<1024, 256, 0, stream>>>(denom, agg, xsum, esum, (long)N * 64);
  // K4: node GEMMs (hx, fx)
  node_gemm_kernel<<<200, 256, nodeLds, stream>>>(x0, nW1T, nW2T, nb1, hxb, nb2 + 64,
                                                  hx, fx, N);
  // K5: edge block + scatter
  edge_kernel<<<512, 256, edgeLds, stream>>>(x0, ei, e0, eW1T, eW2T, eb1, b2e, fx,
                                             e1, denom, agg, esum, E);
  // K6: node finalize
  node_fin_kernel<<<256, 256, 0, stream>>>(x0, hx, agg, denom, x1, xsum, N);
  // K7: global finalize
  gfin_kernel<<<1, 256, 0, stream>>>(g0, xsum, esum, gw1, gb1, gw2, ggb, g1,
                                     1.0f / (float)N, 1.0f / (float)E);
}
